// AttnBlock_32177894981964
// MI455X (gfx1250) — matrix operand order repaired and checked by reading, not compiled, not run on hardware
//
#include <hip/hip_runtime.h>
#include <hip/hip_fp16.h>

#define B_      2
#define C_      512
#define N_      4096   // 64*64
#define GROUPS_ 32
#define CPG_    16     // channels per group
#define EPS_    1e-6f

typedef __attribute__((ext_vector_type(16))) _Float16 v16h;
typedef __attribute__((ext_vector_type(8)))  _Float16 v8h;
typedef __attribute__((ext_vector_type(4)))  _Float16 v4h;
typedef __attribute__((ext_vector_type(8)))  float    v8f;
typedef __attribute__((ext_vector_type(4)))  float    v4f;

#define LDSW 40   // padded row stride (halves) for 32-wide K tiles: 80B, 16B-aligned
#define BT_  128  // block tile (M and N) for the big attention GEMMs

// ---------------------------------------------------------------------------
// Kernel 1: GroupNorm -> f16 activations h16[b][c][n]
// ---------------------------------------------------------------------------
__global__ void gn_kernel(const float* __restrict__ x, const float* __restrict__ gamma,
                          const float* __restrict__ beta, _Float16* __restrict__ h16) {
  const int bg = blockIdx.x;               // B*GROUPS blocks
  const int b = bg / GROUPS_;
  const int g = bg % GROUPS_;
  const int t = threadIdx.x;               // 256 threads
  const int M = CPG_ * N_;                 // 65536 elements per (b, group)
  const float* xp = x + ((size_t)b * C_ + (size_t)g * CPG_) * N_;

  float s = 0.f, ss = 0.f;
  for (int i = t; i < M; i += 256) {
    float v = xp[i];
    s += v; ss += v * v;
  }
  __shared__ float rs[256], rss[256];
  rs[t] = s; rss[t] = ss;
  __syncthreads();
  for (int o = 128; o > 0; o >>= 1) {
    if (t < o) { rs[t] += rs[t + o]; rss[t] += rss[t + o]; }
    __syncthreads();
  }
  const float mean = rs[0] * (1.f / M);
  const float var  = rss[0] * (1.f / M) - mean * mean;
  const float inv  = rsqrtf(var + EPS_);

  _Float16* hp = h16 + ((size_t)b * C_ + (size_t)g * CPG_) * N_;
  for (int i = t; i < M; i += 256) {
    int c = g * CPG_ + (i >> 12);           // i / N_
    float v = (xp[i] - mean) * inv;
    hp[i] = (_Float16)(v * gamma[c] + beta[c]);
  }
}

// ---------------------------------------------------------------------------
// WMMA fragment loaders (wave32, 16x16x32 f16)
// A 16x32: lane<16 -> M=lane, K 0..7 & 16..23 ; lane>=16 -> M=lane-16, K 8..15 & 24..31
// B 32x16: lane<16 -> N=lane, K 0..7 & 16..23 ; lane>=16 -> N=lane-16, K 8..15 & 24..31
// ---------------------------------------------------------------------------
__device__ __forceinline__ v16h load_frag_a(const _Float16* lds, int rowbase, int lane) {
  const int row = rowbase + (lane & 15);
  const int kb = (lane < 16) ? 0 : 8;
  const _Float16* p = lds + row * LDSW + kb;
  v16h f;
  *(v8h*)&f       = *(const v8h*)p;          // K kb..kb+7
  *((v8h*)&f + 1) = *(const v8h*)(p + 16);   // K kb+16..kb+23
  return f;
}

__device__ __forceinline__ v16h load_frag_b(const _Float16* lds, int colbase, int lane) {
  const int col = colbase + (lane & 15);
  const int kb = (lane < 16) ? 0 : 8;
  const _Float16* p = lds + col * LDSW + kb;
  v16h f;
  *(v8h*)&f       = *(const v8h*)p;          // K kb..kb+7
  *((v8h*)&f + 1) = *(const v8h*)(p + 16);   // K kb+16..kb+23
  return f;
}

__device__ __forceinline__ void mma_step(const _Float16* ldsA, const _Float16* ldsB,
                                         int wr, int wc, int lane, v8f acc[2][2]) {
  v16h a0 = load_frag_a(ldsA, wr * 32,      lane);
  v16h a1 = load_frag_a(ldsA, wr * 32 + 16, lane);
  v16h b0 = load_frag_b(ldsB, wc * 32,      lane);
  v16h b1 = load_frag_b(ldsB, wc * 32 + 16, lane);
  acc[0][0] = __builtin_amdgcn_wmma_f32_16x16x32_f16(false, a0, false, b0, (short)0, acc[0][0], false, false);
  acc[0][1] = __builtin_amdgcn_wmma_f32_16x16x32_f16(false, a0, false, b1, (short)0, acc[0][1], false, false);
  acc[1][0] = __builtin_amdgcn_wmma_f32_16x16x32_f16(false, a1, false, b0, (short)0, acc[1][0], false, false);
  acc[1][1] = __builtin_amdgcn_wmma_f32_16x16x32_f16(false, a1, false, b1, (short)0, acc[1][1], false, false);
}

// 8-wave variant: wave computes 32 rows x 64 cols (acc[2][4])
__device__ __forceinline__ void mma_step_2x4(const _Float16* ldsA, const _Float16* ldsB,
                                             int wr, int wc, int lane, v8f acc[2][4]) {
  v16h a[2], bf[4];
#pragma unroll
  for (int i = 0; i < 2; ++i) a[i] = load_frag_a(ldsA, wr * 32 + i * 16, lane);
#pragma unroll
  for (int j = 0; j < 4; ++j) bf[j] = load_frag_b(ldsB, wc * 64 + j * 16, lane);
#pragma unroll
  for (int i = 0; i < 2; ++i)
#pragma unroll
    for (int j = 0; j < 4; ++j)
      acc[i][j] = __builtin_amdgcn_wmma_f32_16x16x32_f16(false, a[i], false, bf[j], (short)0,
                                                         acc[i][j], false, false);
}

// ---------------------------------------------------------------------------
// LDS staging helpers
// ---------------------------------------------------------------------------
// f32 weight tile (row stride C_) -> f16 LDS, natural [row][k], 64x32, 128 thr
__device__ __forceinline__ void stage_tile_w(_Float16* lds, const float* g, int t) {
#pragma unroll
  for (int i = 0; i < 4; ++i) {
    int e4 = t + i * 128;
    int row = e4 >> 3;
    int c4 = (e4 & 7) * 4;
    v4f wv = *(const v4f*)(g + (size_t)row * C_ + c4);
    v4h hv = { (_Float16)wv.x, (_Float16)wv.y, (_Float16)wv.z, (_Float16)wv.w };
    *(v4h*)(lds + row * LDSW + c4) = hv;
  }
}

// f16 tile natural copy: 64 rows x 32 halves, 128 thr
__device__ __forceinline__ void stage_tile_f16(_Float16* lds, const _Float16* g,
                                               size_t ld, int t) {
#pragma unroll
  for (int i = 0; i < 2; ++i) {
    int e8 = t + i * 128;
    int row = e8 >> 2;
    int c8 = (e8 & 3) * 8;
    *(v8h*)(lds + row * LDSW + c8) = *(const v8h*)(g + (size_t)row * ld + c8);
  }
}

// f16 tile transposing copy: global 32 rows (k) x 64 cols (n) -> lds[n][k], 128 thr
__device__ __forceinline__ void stage_tile_f16_T(_Float16* lds, const _Float16* g,
                                                 size_t ld, int t) {
#pragma unroll
  for (int i = 0; i < 2; ++i) {
    int e8 = t + i * 128;
    int kk = e8 >> 3;
    int nn = (e8 & 7) * 8;
    v8h hv = *(const v8h*)(g + (size_t)kk * ld + nn);
#pragma unroll
    for (int j = 0; j < 8; ++j) lds[(nn + j) * LDSW + kk] = hv[j];
  }
}

// CDNA5 async copy: 128 rows x 32 halves tile, 256 thr, 2 async ops per thread.
// GLOBAL_LOAD_ASYNC_TO_LDS_B128: per-lane LDS byte address (low 32 bits of the
// flat LDS aperture address) + per-lane 64-bit global address; tracked by ASYNCcnt.
__device__ __forceinline__ void stage_async_128x32(_Float16* lds, const _Float16* g,
                                                   size_t ld, int t) {
#pragma unroll
  for (int i = 0; i < 2; ++i) {
    int e8 = t + i * 256;                 // 512 chunks of 8 halves (4 per row)
    int row = e8 >> 2;
    int c8 = (e8 & 3) * 8;
    unsigned loff = (unsigned)(size_t)(lds + row * LDSW + c8);
    const _Float16* gp = g + (size_t)row * ld + c8;
    asm volatile("global_load_async_to_lds_b128 %0, %1, off"
                 :: "v"(loff), "v"(gp) : "memory");
  }
}

// ---------------------------------------------------------------------------
// Kernel 2: QKV projection. out = W(CxC) * h16(CxN) + bias  (64x64 tile, 128 thr)
// TRANS_OUT=1: store f16 [N][C] (for Q^T, K^T); else f16 [C][N] (for V)
// ---------------------------------------------------------------------------
template <bool TRANS_OUT>
__global__ void gemm_qkv_kernel(const float* __restrict__ Wm, const float* __restrict__ bias,
                                const _Float16* __restrict__ Hm, _Float16* __restrict__ out) {
  const int n0 = blockIdx.x * 64;
  const int m0 = blockIdx.y * 64;
  const int b  = blockIdx.z;
  const int t = threadIdx.x, wave = t >> 5, lane = t & 31;
  const int wr = wave >> 1, wc = wave & 1;
  __shared__ _Float16 ldsA[64 * LDSW];
  __shared__ _Float16 ldsB[64 * LDSW];
  const _Float16* Hb = Hm + (size_t)b * C_ * N_;

  v8f acc[2][2] = {};
  for (int k0 = 0; k0 < C_; k0 += 32) {
    stage_tile_w(ldsA, Wm + (size_t)m0 * C_ + k0, t);
    stage_tile_f16_T(ldsB, Hb + (size_t)k0 * N_ + n0, N_, t);
    __syncthreads();
    mma_step(ldsA, ldsB, wr, wc, lane, acc);
    __syncthreads();
  }

  const int mb = (lane < 16) ? 0 : 8;
  const int lc = lane & 15;
#pragma unroll
  for (int i = 0; i < 2; ++i) {
    const int mrow0 = m0 + wr * 32 + i * 16 + mb;
#pragma unroll
    for (int j = 0; j < 2; ++j) {
      const int ncol = n0 + wc * 32 + j * 16 + lc;
      if (TRANS_OUT) {
        v8h hv;
#pragma unroll
        for (int r = 0; r < 8; ++r)
          hv[r] = (_Float16)(acc[i][j][r] + bias[mrow0 + r]);
        *(v8h*)(out + (size_t)b * N_ * C_ + (size_t)ncol * C_ + mrow0) = hv;
      } else {
#pragma unroll
        for (int r = 0; r < 8; ++r)
          out[(size_t)b * C_ * N_ + (size_t)(mrow0 + r) * N_ + ncol] =
              (_Float16)(acc[i][j][r] + bias[mrow0 + r]);
      }
    }
  }
}

// ---------------------------------------------------------------------------
// Kernel 3: scores[nq][nk] = scale * sum_c qT[nq][c] * kT[nk][c]   (f32 out)
// 128x128 block tile, 256 thr, double-buffered async-to-LDS staging.
// ---------------------------------------------------------------------------
__global__ void __launch_bounds__(256)
gemm_scores_kernel(const _Float16* __restrict__ qT, const _Float16* __restrict__ kT,
                   float* __restrict__ S) {
  const int nk0 = blockIdx.x * BT_;
  const int nq0 = blockIdx.y * BT_;
  const int b   = blockIdx.z;
  const int t = threadIdx.x, wave = t >> 5, lane = t & 31;
  const int wr = wave >> 1, wc = wave & 1;       // 4x2 wave grid
  __shared__ _Float16 lds[2][2][BT_ * LDSW];     // [buf][A/B], 40 KB total
  const _Float16* Qt = qT + (size_t)b * N_ * C_ + (size_t)nq0 * C_;
  const _Float16* Kt = kT + (size_t)b * N_ * C_ + (size_t)nk0 * C_;

  v8f acc[2][4] = {};
  stage_async_128x32(lds[0][0], Qt, C_, t);
  stage_async_128x32(lds[0][1], Kt, C_, t);
  const int T = C_ / 32;                         // 16 K-steps
  for (int it = 0; it < T; ++it) {
    const int cur = it & 1;
    if (it + 1 < T) {
      stage_async_128x32(lds[cur ^ 1][0], Qt + (it + 1) * 32, C_, t);
      stage_async_128x32(lds[cur ^ 1][1], Kt + (it + 1) * 32, C_, t);
      // 4 async instrs per wave per tile; in-order completion => remaining 4
      // outstanding are next tile's, so current tile is fully in LDS.
      asm volatile("s_wait_asynccnt 0x4" ::: "memory");
    } else {
      asm volatile("s_wait_asynccnt 0x0" ::: "memory");
    }
    __syncthreads();
    mma_step_2x4(lds[cur][0], lds[cur][1], wr, wc, lane, acc);
    __syncthreads();
  }

  const float scale = 0.04419417382415922f;      // 512^-0.5
  const int mb = (lane < 16) ? 0 : 8;
  const int lc = lane & 15;
  float* Sb = S + (size_t)b * N_ * N_;
#pragma unroll
  for (int i = 0; i < 2; ++i) {
    const int q0 = nq0 + wr * 32 + i * 16 + mb;
#pragma unroll
    for (int j = 0; j < 4; ++j) {
      const int kcol = nk0 + wc * 64 + j * 16 + lc;
#pragma unroll
      for (int r = 0; r < 8; ++r)
        Sb[(size_t)(q0 + r) * N_ + kcol] = acc[i][j][r] * scale;
    }
  }
}

// ---------------------------------------------------------------------------
// Kernel 4: row softmax (over keys), f32 scores -> f16 attn
// ---------------------------------------------------------------------------
__global__ void softmax_kernel(const float* __restrict__ S, _Float16* __restrict__ A) {
  const size_t row = blockIdx.x;               // B*N rows
  const float* sp = S + row * N_;
  _Float16* ap = A + row * N_;
  const int t = threadIdx.x;                   // 256
  __shared__ float red[256];

  float m = -3.4e38f;
  for (int i = t; i < N_; i += 256) m = fmaxf(m, sp[i]);
  red[t] = m; __syncthreads();
  for (int o = 128; o > 0; o >>= 1) {
    if (t < o) red[t] = fmaxf(red[t], red[t + o]);
    __syncthreads();
  }
  m = red[0];
  __syncthreads();

  float s = 0.f;
  for (int i = t; i < N_; i += 256) {
    float e = __expf(sp[i] - m);
    s += e;
    ap[i] = (_Float16)e;
  }
  red[t] = s; __syncthreads();
  for (int o = 128; o > 0; o >>= 1) {
    if (t < o) red[t] += red[t + o];
    __syncthreads();
  }
  const float inv = 1.f / red[0];
  for (int i = t; i < N_; i += 256)
    ap[i] = (_Float16)((float)ap[i] * inv);
}

// ---------------------------------------------------------------------------
// Kernel 5: OT[nq][c] = sum_m V[c][m] * attn[nq][m]   (K = 4096)
// 128x128 block tile, 256 thr, double-buffered async staging.
// ---------------------------------------------------------------------------
__global__ void __launch_bounds__(256)
gemm_av_kernel(const _Float16* __restrict__ V, const _Float16* __restrict__ A,
               _Float16* __restrict__ OT) {
  const int n0 = blockIdx.x * BT_;   // query tile
  const int m0 = blockIdx.y * BT_;   // channel tile
  const int b  = blockIdx.z;
  const int t = threadIdx.x, wave = t >> 5, lane = t & 31;
  const int wr = wave >> 1, wc = wave & 1;
  __shared__ _Float16 lds[2][2][BT_ * LDSW];
  const _Float16* Vb = V + (size_t)b * C_ * N_ + (size_t)m0 * N_;   // rows = channels
  const _Float16* Ab = A + (size_t)b * N_ * N_ + (size_t)n0 * N_;   // rows = queries

  v8f acc[2][4] = {};
  stage_async_128x32(lds[0][0], Vb, N_, t);
  stage_async_128x32(lds[0][1], Ab, N_, t);
  const int T = N_ / 32;                         // 128 K-steps
  for (int it = 0; it < T; ++it) {
    const int cur = it & 1;
    if (it + 1 < T) {
      stage_async_128x32(lds[cur ^ 1][0], Vb + (it + 1) * 32, N_, t);
      stage_async_128x32(lds[cur ^ 1][1], Ab + (it + 1) * 32, N_, t);
      asm volatile("s_wait_asynccnt 0x4" ::: "memory");
    } else {
      asm volatile("s_wait_asynccnt 0x0" ::: "memory");
    }
    __syncthreads();
    mma_step_2x4(lds[cur][0], lds[cur][1], wr, wc, lane, acc);
    __syncthreads();
  }

  const int mb = (lane < 16) ? 0 : 8;
  const int lc = lane & 15;
#pragma unroll
  for (int i = 0; i < 2; ++i) {
    const int c0 = m0 + wr * 32 + i * 16 + mb;
#pragma unroll
    for (int j = 0; j < 4; ++j) {
      const int q = n0 + wc * 64 + j * 16 + lc;
      v8h hv;
#pragma unroll
      for (int r = 0; r < 8; ++r) hv[r] = (_Float16)acc[i][j][r];
      *(v8h*)(OT + (size_t)b * N_ * C_ + (size_t)q * C_ + c0) = hv;
    }
  }
}

// ---------------------------------------------------------------------------
// Kernel 6: y = Wo * attn_out + bo + x   (f32 out, natural [C][N] layout)
// ---------------------------------------------------------------------------
__global__ void gemm_out_kernel(const float* __restrict__ Wo, const float* __restrict__ bo,
                                const _Float16* __restrict__ OT, const float* __restrict__ x,
                                float* __restrict__ y) {
  const int n0 = blockIdx.x * 64;
  const int m0 = blockIdx.y * 64;
  const int b  = blockIdx.z;
  const int t = threadIdx.x, wave = t >> 5, lane = t & 31;
  const int wr = wave >> 1, wc = wave & 1;
  __shared__ _Float16 ldsA[64 * LDSW];
  __shared__ _Float16 ldsB[64 * LDSW];
  const _Float16* Ob = OT + (size_t)b * N_ * C_;

  v8f acc[2][2] = {};
  for (int k0 = 0; k0 < C_; k0 += 32) {
    stage_tile_w(ldsA, Wo + (size_t)m0 * C_ + k0, t);
    stage_tile_f16(ldsB, Ob + (size_t)n0 * C_ + k0, C_, t);   // rows = pixels
    __syncthreads();
    mma_step(ldsA, ldsB, wr, wc, lane, acc);
    __syncthreads();
  }

  const int mb = (lane < 16) ? 0 : 8;
  const int lc = lane & 15;
#pragma unroll
  for (int i = 0; i < 2; ++i) {
    const int c0 = m0 + wr * 32 + i * 16 + mb;
#pragma unroll
    for (int j = 0; j < 2; ++j) {
      const int n = n0 + wc * 32 + j * 16 + lc;
#pragma unroll
      for (int r = 0; r < 8; ++r) {
        size_t idx = ((size_t)b * C_ + (size_t)(c0 + r)) * N_ + n;
        y[idx] = acc[i][j][r] + bo[c0 + r] + x[idx];
      }
    }
  }
}

// ---------------------------------------------------------------------------
extern "C" void kernel_launch(void* const* d_in, const int* in_sizes, int n_in,
                              void* d_out, int out_size, void* d_ws, size_t ws_size,
                              hipStream_t stream) {
  const float* x  = (const float*)d_in[0];
  const float* gg = (const float*)d_in[1];
  const float* gb = (const float*)d_in[2];
  const float* wq = (const float*)d_in[3];
  const float* bq = (const float*)d_in[4];
  const float* wk = (const float*)d_in[5];
  const float* bk = (const float*)d_in[6];
  const float* wv = (const float*)d_in[7];
  const float* bv = (const float*)d_in[8];
  const float* wo = (const float*)d_in[9];
  const float* bo = (const float*)d_in[10];
  float* out = (float*)d_out;
  (void)in_sizes; (void)n_in; (void)out_size; (void)ws_size;

  char* ws = (char*)d_ws;
  size_t off = 0;
  auto alloc = [&](size_t bytes) -> void* {
    void* p = ws + off;
    off += (bytes + 255) & ~(size_t)255;
    return p;
  };
  _Float16* h16 = (_Float16*)alloc((size_t)B_ * C_ * N_ * 2);   // GN output  [b][c][n]
  _Float16* qT  = (_Float16*)alloc((size_t)B_ * N_ * C_ * 2);   // Q^T        [b][n][c]
  _Float16* kT  = (_Float16*)alloc((size_t)B_ * N_ * C_ * 2);   // K^T        [b][n][c]
  _Float16* v16 = (_Float16*)alloc((size_t)B_ * C_ * N_ * 2);   // V          [b][c][n]
  float*    S   = (float*)   alloc((size_t)B_ * N_ * N_ * 4);   // scores     [b][nq][nk]
  _Float16* A16 = (_Float16*)alloc((size_t)B_ * N_ * N_ * 2);   // attn       [b][nq][nk]
  _Float16* OT  = (_Float16*)alloc((size_t)B_ * N_ * C_ * 2);   // attn out^T [b][n][c]

  gn_kernel<<<B_ * GROUPS_, 256, 0, stream>>>(x, gg, gb, h16);

  dim3 gQKV(N_ / 64, C_ / 64, B_);
  gemm_qkv_kernel<true ><<<gQKV, 128, 0, stream>>>(wq, bq, h16, qT);
  gemm_qkv_kernel<true ><<<gQKV, 128, 0, stream>>>(wk, bk, h16, kT);
  gemm_qkv_kernel<false><<<gQKV, 128, 0, stream>>>(wv, bv, h16, v16);

  dim3 gS(N_ / BT_, N_ / BT_, B_);
  gemm_scores_kernel<<<gS, 256, 0, stream>>>(qT, kT, S);

  softmax_kernel<<<B_ * N_, 256, 0, stream>>>(S, A16);

  dim3 gAV(N_ / BT_, C_ / BT_, B_);
  gemm_av_kernel<<<gAV, 256, 0, stream>>>(v16, A16, OT);

  dim3 gO(N_ / 64, C_ / 64, B_);
  gemm_out_kernel<<<gO, 128, 0, stream>>>(wo, bo, OT, x, out);
}
